// Fastattention_24704651887079
// MI455X (gfx1250) — matrix operand order repaired and checked by reading, not compiled, not run on hardware
//
#include <hip/hip_runtime.h>
#include <stdint.h>

typedef __bf16 bf16;
typedef __attribute__((ext_vector_type(16))) __bf16 v16bf;
typedef __attribute__((ext_vector_type(8)))  float  v8f;

static constexpr int B = 8, T = 4096, E = 512, H = 8, Dh = 64;

union FragAB { v16bf v; uint4 q[2]; };

// -------------------- conversion kernels --------------------
__global__ void cvt_f32_to_bf16(const float* __restrict__ src, bf16* __restrict__ dst, long n) {
  long i = (long)blockIdx.x * blockDim.x + threadIdx.x;
  if (i < n) dst[i] = (bf16)src[i];
}

// Wt[n*E + k] = W[k*E + n]  (store weight transposed: [N][K], bf16)
__global__ void transpose_w_bf16(const float* __restrict__ W, bf16* __restrict__ Wt) {
  int i = blockIdx.x * blockDim.x + threadIdx.x;  // E*E threads
  int n = i / E, k = i % E;
  Wt[(long)n * E + k] = (bf16)W[(long)k * E + n];
}

// -------------------- WMMA GEMM --------------------
// C[b] = A[b] (MxK, bf16 row-major) @ Bt[b]^T (Bt is [N][K] bf16) + bias + residual
// Block tile: 128(M) x 256(N); 8 waves, each wave a 64x64 tile (4x4 WMMA frags).
__global__ __launch_bounds__(256) void wmma_gemm_bf16(
    const bf16* __restrict__ A, const bf16* __restrict__ Bt,
    const float* __restrict__ bias, const float* __restrict__ residual,
    float* __restrict__ Cf, bf16* __restrict__ Ch,
    int N, int K, long sA, long sB, long sC)
{
  const int bz = blockIdx.z;
  A  += (long)bz * sA;
  Bt += (long)bz * sB;
  const long cbase = (long)bz * sC;

  const int lane = threadIdx.x & 31;
  const int wid  = threadIdx.x >> 5;
  const int lr   = lane & 15;   // row (A) / col (B,C) within fragment
  const int lhi  = lane >> 4;   // K-half select (A,B); M-half select (C)

  const int wm0 = blockIdx.y * 128 + (wid >> 2) * 64;
  const int wn0 = blockIdx.x * 256 + (wid & 3) * 64;

  v8f zero = {};
  v8f acc[4][4];
  #pragma unroll
  for (int i = 0; i < 4; ++i)
    #pragma unroll
    for (int j = 0; j < 4; ++j) acc[i][j] = zero;

  for (int k0 = 0; k0 < K; k0 += 32) {
    if (k0 + 32 < K)  // hint next K tile of A into cache (global_prefetch_b8)
      __builtin_prefetch(A + (long)(wm0 + lr) * K + k0 + 32, 0, 1);

    v16bf af[4], bfr[4];
    #pragma unroll
    for (int tm = 0; tm < 4; ++tm) {
      // A 16x32 bf16 frag: lane holds row m=lr; elems 0..7 = K(k0+lhi*8 ..+7),
      // elems 8..15 = K(k0+lhi*8+16 ..+23)  (ISA 16-bit A layout)
      FragAB f;
      const bf16* p = A + (long)(wm0 + tm * 16 + lr) * K + k0 + lhi * 8;
      f.q[0] = *(const uint4*)(p);
      f.q[1] = *(const uint4*)(p + 16);
      af[tm] = f.v;
    }
    #pragma unroll
    for (int tn = 0; tn < 4; ++tn) {
      // B 32x16 bf16 frag: lane holds col n=lr; elems 0..7 = K(k0+lhi*8 ..+7),
      // elems 8..15 = K(k0+lhi*8+16 ..+23)
      FragAB f;
      const bf16* p = Bt + (long)(wn0 + tn * 16 + lr) * K + k0 + lhi * 8;
      f.q[0] = *(const uint4*)(p);
      f.q[1] = *(const uint4*)(p + 16);
      bfr[tn] = f.v;
    }
    #pragma unroll
    for (int tm = 0; tm < 4; ++tm)
      #pragma unroll
      for (int tn = 0; tn < 4; ++tn)
        acc[tm][tn] = __builtin_amdgcn_wmma_f32_16x16x32_bf16(
            false, af[tm], false, bfr[tn], (short)0, acc[tm][tn], false, false);
  }

  // C 16x16 f32 frag: elem r -> row m = r + 8*lhi, col n = lr
  #pragma unroll
  for (int tm = 0; tm < 4; ++tm) {
    #pragma unroll
    for (int tn = 0; tn < 4; ++tn) {
      const int n = wn0 + tn * 16 + lr;
      const float bn = bias ? bias[n] : 0.0f;
      #pragma unroll
      for (int r = 0; r < 8; ++r) {
        const int m = wm0 + tm * 16 + lhi * 8 + r;
        const long idx = cbase + (long)m * N + n;
        float c = acc[tm][tn][r] + bn;
        if (residual) c += residual[idx];
        Cf[idx] = c;
        if (Ch) Ch[idx] = (bf16)c;
      }
    }
  }
}

// -------------------- score GEMM (N = H = 8, plain VALU) --------------------
// out[row,h] = sum_e X[row,e] * Wa[(b?)e,h] + bias[h]; 32 rows per 256-thread block.
__global__ __launch_bounds__(256) void scores_kernel(
    const float* __restrict__ X, const float* __restrict__ Wa,
    const float* __restrict__ bias, float* __restrict__ out, int perBatchW)
{
  const int tid = threadIdx.x;
  const int h = tid & (H - 1);
  const int row = blockIdx.x * 32 + (tid >> 3);
  const int b = row / T;
  const float* wp = Wa + (perBatchW ? (long)b * E * H : 0);
  const float* xr = X + (long)row * E;
  float acc = bias[h];
  for (int e = 0; e < E; ++e)
    acc = fmaf(xr[e], wp[e * H + h], acc);
  out[(long)row * H + h] = acc;
}

// -------------------- masked softmax-pool over T --------------------
// One block per (h, b). out[b, h*64+d] = sum_t softmax_t(scores/8, mask) * X[b,t,h*64+d]
// (optionally * scale[b, h*64+d]).
__global__ __launch_bounds__(256) void pool_kernel(
    const float* __restrict__ scores, const unsigned char* __restrict__ mask,
    const float* __restrict__ X, const float* __restrict__ scale,
    float* __restrict__ out)
{
  const int h = blockIdx.x, b = blockIdx.y, tid = threadIdx.x;
  __shared__ float red[256];
  const float* sc = scores + (long)b * T * H + h;     // stride H over t
  const unsigned char* mk = mask + (long)b * T;
  const float kscale = 0.125f;                        // 1/sqrt(64)

  float mx = -__builtin_inff();
  for (int t = tid; t < T; t += 256)
    if (!mk[t]) mx = fmaxf(mx, sc[(long)t * H] * kscale);
  red[tid] = mx; __syncthreads();
  for (int off = 128; off > 0; off >>= 1) {
    if (tid < off) red[tid] = fmaxf(red[tid], red[tid + off]);
    __syncthreads();
  }
  mx = red[0]; __syncthreads();

  float sm = 0.0f;
  for (int t = tid; t < T; t += 256)
    if (!mk[t]) sm += __expf(sc[(long)t * H] * kscale - mx);
  red[tid] = sm; __syncthreads();
  for (int off = 128; off > 0; off >>= 1) {
    if (tid < off) red[tid] += red[tid + off];
    __syncthreads();
  }
  const float inv = 1.0f / red[0]; __syncthreads();

  const int d = tid & 63, tg = tid >> 6;
  float acc = 0.0f;
  const float* xb = X + (long)b * T * E + h * Dh + d;
  for (int t = tg; t < T; t += 4)
    if (!mk[t]) {
      float w = __expf(sc[(long)t * H] * kscale - mx) * inv;
      acc = fmaf(w, xb[(long)t * E], acc);
    }
  red[tid] = acc; __syncthreads();
  if (tg == 0) {
    float r = red[d] + red[64 + d] + red[128 + d] + red[192 + d];
    const int e = h * Dh + d;
    if (scale) r *= scale[(long)b * E + e];
    out[(long)b * E + e] = r;
  }
}

// -------------------- weight-prep fusions --------------------
// WakB[b,e,h] = q_flat[b,e] * Wak[e,h]   (folds key_content into per-batch weight)
__global__ void prep_wak(const float* __restrict__ qflat, const float* __restrict__ Wak,
                         float* __restrict__ out) {
  int i = blockIdx.x * blockDim.x + threadIdx.x;   // B*E*H
  int b = i / (E * H), r = i % (E * H), e = r / H;
  out[i] = qflat[b * E + e] * Wak[r];
}

// WoTb[b][n][k] = bf16( s[b,k] * Wo[k,n] )   (folds val = query * s into Wo, transposed)
__global__ void prep_wo(const float* __restrict__ s, const float* __restrict__ Wo,
                        bf16* __restrict__ WoTb) {
  long i = (long)blockIdx.x * blockDim.x + threadIdx.x;  // B*E*E
  int b = (int)(i / ((long)E * E));
  int r = (int)(i % ((long)E * E));
  int n = r / E, k = r % E;
  WoTb[i] = (bf16)(s[(long)b * E + k] * Wo[(long)k * E + n]);
}

// -------------------- launch --------------------
extern "C" void kernel_launch(void* const* d_in, const int* in_sizes, int n_in,
                              void* d_out, int out_size, void* d_ws, size_t ws_size,
                              hipStream_t stream)
{
  const float* x    = (const float*)d_in[0];          // q (k, v unused by module)
  const unsigned char* mask = (const unsigned char*)d_in[3];  // bool, 1 byte/elem
  const float* Wq  = (const float*)d_in[4];
  const float* bq  = (const float*)d_in[5];
  const float* Waq = (const float*)d_in[6];
  const float* baq = (const float*)d_in[7];
  const float* Wk  = (const float*)d_in[8];
  const float* bk  = (const float*)d_in[9];
  const float* Wak = (const float*)d_in[10];
  const float* bak = (const float*)d_in[11];
  const float* Wo  = (const float*)d_in[12];
  const float* bo  = (const float*)d_in[13];
  float* out = (float*)d_out;
  (void)in_sizes; (void)n_in; (void)out_size; (void)ws_size;

  char* ws = (char*)d_ws;
  auto alloc = [&](size_t bytes) -> char* {
    char* p = ws; ws += (bytes + 255) & ~(size_t)255; return p;
  };
  bf16*  xh    = (bf16*) alloc((size_t)B * T * E * 2);   // x in bf16
  bf16*  WqT   = (bf16*) alloc((size_t)E * E * 2);       // Wq^T bf16
  bf16*  WkT   = (bf16*) alloc((size_t)E * E * 2);       // Wk^T bf16
  float* qf    = (float*)alloc((size_t)B * T * E * 4);   // query fp32
  bf16*  qh    = (bf16*) alloc((size_t)B * T * E * 2);   // query bf16
  float* kf    = (float*)alloc((size_t)B * T * E * 4);   // key fp32
  float* qsc   = (float*)alloc((size_t)B * T * H * 4);   // q_scores
  float* ksc   = (float*)alloc((size_t)B * T * H * 4);   // k_scores
  float* qflat = (float*)alloc((size_t)B * E * 4);       // pooled q summary
  float* sflat = (float*)alloc((size_t)B * E * 4);       // k_sum * q_flat
  float* wakb  = (float*)alloc((size_t)B * E * H * 4);   // per-batch Wak'
  bf16*  woTb  = (bf16*) alloc((size_t)B * E * E * 2);   // per-batch diag(s)@Wo, transposed

  const long n_x = (long)B * T * E;
  cvt_f32_to_bf16<<<dim3((unsigned)((n_x + 255) / 256)), 256, 0, stream>>>(x, xh, n_x);
  transpose_w_bf16<<<dim3(E * E / 256), 256, 0, stream>>>(Wq, WqT);
  transpose_w_bf16<<<dim3(E * E / 256), 256, 0, stream>>>(Wk, WkT);

  dim3 g1(E / 256, (B * T) / 128, 1);
  wmma_gemm_bf16<<<g1, 256, 0, stream>>>(xh, WqT, bq, nullptr, qf, qh, E, E, 0, 0, 0);
  wmma_gemm_bf16<<<g1, 256, 0, stream>>>(xh, WkT, bk, nullptr, kf, nullptr, E, E, 0, 0, 0);

  scores_kernel<<<dim3(B * T / 32), 256, 0, stream>>>(qf, Waq, baq, qsc, 0);
  pool_kernel<<<dim3(H, B), 256, 0, stream>>>(qsc, mask, qf, nullptr, qflat);
  prep_wak<<<dim3(B * E * H / 256), 256, 0, stream>>>(qflat, Wak, wakb);
  scores_kernel<<<dim3(B * T / 32), 256, 0, stream>>>(kf, wakb, bak, ksc, 1);
  pool_kernel<<<dim3(H, B), 256, 0, stream>>>(ksc, mask, kf, qflat, sflat);
  prep_wo<<<dim3((unsigned)((long)B * E * E / 256)), 256, 0, stream>>>(sflat, Wo, woTb);

  // out[b] = query_bf16[b] @ (diag(s_b)@Wo) + bo + query_f32[b]
  dim3 g2(E / 256, T / 128, B);
  wmma_gemm_bf16<<<g2, 256, 0, stream>>>(qh, woTb, bo, qf, out, nullptr,
                                         E, E, (long)T * E, (long)E * E, (long)T * E);
}
